// TransformerEncoderBlock_32813550141932
// MI455X (gfx1250) — compile-verified, matrix operand order checked by reading, not run on hardware
//
#include <hip/hip_runtime.h>
#include <math.h>

// ---------------------------------------------------------------------------
// Types for CDNA5 WMMA (wave32): 16x16x32 f16 -> f32 accumulate
// ---------------------------------------------------------------------------
typedef _Float16 h16;
typedef __attribute__((ext_vector_type(8)))  h16   v8h;
typedef __attribute__((ext_vector_type(16))) h16   v16h;
typedef __attribute__((ext_vector_type(8)))  float v8f;

#define D_MODEL 1024
#define N_HEADS 16
#define DK      64
#define DIM_FF  4096
#define BATCH   2
#define SEQ     2048
#define M_ROWS  (BATCH * SEQ) /* 4096 */

__device__ __forceinline__ v8f wmma_f16(v16h a, v16h b, v8f c) {
  // (neg_a, A, neg_b, B, c_mod, C, reuse_a, reuse_b)
  return __builtin_amdgcn_wmma_f32_16x16x32_f16(false, a, false, b,
                                                (short)0, c, false, false);
}

// Load one 16x32 (or 32x16 col-major) f16 fragment per ISA layout:
// lanes 0-15: elems[0..7]=K(k0..k0+7), elems[8..15]=K(k0+16..k0+23)
// lanes16-31: elems[0..7]=K(k0+8..k0+15), elems[8..15]=K(k0+24..k0+31)
__device__ __forceinline__ v16h load_frag(const h16* __restrict__ base,
                                          int row, int ld, int k0, int hlf) {
  const h16* p = base + (size_t)row * ld + k0 + hlf * 8;
  v8h lo = *reinterpret_cast<const v8h*>(p);
  v8h hi = *reinterpret_cast<const v8h*>(p + 16);
  v16h f;
#pragma unroll
  for (int i = 0; i < 8; ++i) { f[i] = lo[i]; f[i + 8] = hi[i]; }
  return f;
}

// ---------------------------------------------------------------------------
// Weight prep: f32 [R][C] -> f16 transposed [C][R]
// ---------------------------------------------------------------------------
__global__ __launch_bounds__(256) void transpose_f32_to_f16(
    const float* __restrict__ in, h16* __restrict__ out, int R, int C) {
  size_t idx = (size_t)blockIdx.x * 256 + threadIdx.x;
  if (idx < (size_t)R * C) {
    int r = (int)(idx / C);
    int c = (int)(idx % C);
    out[(size_t)c * R + r] = (h16)in[idx];
  }
}

// ---------------------------------------------------------------------------
// LayerNorm: f32 row (D_MODEL) -> f16 row (feeds WMMA GEMMs)
// ---------------------------------------------------------------------------
__global__ __launch_bounds__(256) void layernorm_to_f16(
    const float* __restrict__ x, const float* __restrict__ gamma,
    const float* __restrict__ beta, h16* __restrict__ out) {
  __shared__ float s1[256];
  __shared__ float s2[256];
  const int row = blockIdx.x;
  const int tid = threadIdx.x;
  const float* xr = x + (size_t)row * D_MODEL;
  float a = 0.f, b = 0.f;
#pragma unroll
  for (int i = 0; i < 4; ++i) {
    float v = xr[tid + 256 * i];
    a += v;
    b += v * v;
  }
  s1[tid] = a;
  s2[tid] = b;
  __syncthreads();
  for (int off = 128; off > 0; off >>= 1) {
    if (tid < off) { s1[tid] += s1[tid + off]; s2[tid] += s2[tid + off]; }
    __syncthreads();
  }
  const float mu  = s1[0] * (1.0f / D_MODEL);
  const float var = s2[0] * (1.0f / D_MODEL) - mu * mu;
  const float inv = rsqrtf(var + 1e-5f);
#pragma unroll
  for (int i = 0; i < 4; ++i) {
    int c = tid + 256 * i;
    out[(size_t)row * D_MODEL + c] = (h16)((xr[c] - mu) * inv * gamma[c] + beta[c]);
  }
}

// ---------------------------------------------------------------------------
// WMMA GEMM: C = epilogue(A[M,K] @ Bt[N,K]^T + bias)
// 256 threads = 8 waves (2x4); each wave computes a 32x32 tile.
// ---------------------------------------------------------------------------
enum { MODE_QKV = 0, MODE_GELU_F16 = 1, MODE_RES_F32 = 2, MODE_V_T = 3 };

template <int MODE>
__global__ __launch_bounds__(256) void gemm_f16_wmma(
    const h16* __restrict__ A, const h16* __restrict__ Bt,
    const float* __restrict__ bias, const float* __restrict__ residual,
    void* __restrict__ outv, int M, int N, int K) {
  const int tid  = threadIdx.x;
  const int wave = tid >> 5;
  const int lane = tid & 31;
  const int wm   = wave >> 2;  // 0..1
  const int wn   = wave & 3;   // 0..3
  const int row  = lane & 15;
  const int hlf  = lane >> 4;

  const int m_base = blockIdx.y * 64 + wm * 32;
  const int n_base = blockIdx.x * 128 + wn * 32;

  const h16* pa0 = A + (size_t)(m_base + row) * K;
  const h16* pa1 = A + (size_t)(m_base + 16 + row) * K;
  const h16* pb0 = Bt + (size_t)(n_base + row) * K;
  const h16* pb1 = Bt + (size_t)(n_base + 16 + row) * K;

  v8f acc[2][2] = {};
  for (int k0 = 0; k0 < K; k0 += 32) {
    v16h a0 = load_frag(pa0, 0, K, k0, hlf);
    v16h a1 = load_frag(pa1, 0, K, k0, hlf);
    v16h b0 = load_frag(pb0, 0, K, k0, hlf);
    v16h b1 = load_frag(pb1, 0, K, k0, hlf);
    if (k0 + 64 < K) {  // hide HBM latency behind the WMMA chain
      __builtin_prefetch(pa0 + k0 + 64, 0, 1);
      __builtin_prefetch(pa1 + k0 + 64, 0, 1);
      __builtin_prefetch(pb0 + k0 + 64, 0, 1);
      __builtin_prefetch(pb1 + k0 + 64, 0, 1);
    }
    acc[0][0] = wmma_f16(a0, b0, acc[0][0]);
    acc[0][1] = wmma_f16(a0, b1, acc[0][1]);
    acc[1][0] = wmma_f16(a1, b0, acc[1][0]);
    acc[1][1] = wmma_f16(a1, b1, acc[1][1]);
  }

  // C/D layout: lane 0-15 -> N=lane, M=r; lane 16-31 -> N=lane-16, M=8+r
  const int m0 = hlf * 8;
#pragma unroll
  for (int tm = 0; tm < 2; ++tm) {
#pragma unroll
    for (int tn = 0; tn < 2; ++tn) {
      const int gn = n_base + tn * 16 + row;
      const float bsv = bias[gn];
#pragma unroll
      for (int r = 0; r < 8; ++r) {
        const int gm = m_base + tm * 16 + m0 + r;
        float val = acc[tm][tn][r] + bsv;
        if (MODE == MODE_RES_F32) {
          float* out = (float*)outv;
          out[(size_t)gm * N + gn] = val + residual[(size_t)gm * N + gn];
        } else if (MODE == MODE_GELU_F16) {
          h16* out = (h16*)outv;
          float ge = 0.5f * val * (1.0f + erff(val * 0.70710678118654752f));
          out[(size_t)gm * N + gn] = (h16)ge;
        } else if (MODE == MODE_QKV) {  // store [b][h][t][d] f16
          h16* out = (h16*)outv;
          int b = gm >> 11, t = gm & (SEQ - 1);
          int h = gn >> 6,  d = gn & (DK - 1);
          out[(((size_t)b * N_HEADS + h) * SEQ + t) * DK + d] = (h16)val;
        } else {  // MODE_V_T: store [b][h][d][t] f16 (K-major for P@V WMMA)
          h16* out = (h16*)outv;
          int b = gm >> 11, t = gm & (SEQ - 1);
          int h = gn >> 6,  d = gn & (DK - 1);
          out[(((size_t)b * N_HEADS + h) * DK + d) * SEQ + t] = (h16)val;
        }
      }
    }
  }
}

// ---------------------------------------------------------------------------
// Fused attention: one WG per (b,h, 16-query tile). 4 waves, 128 threads.
// Scores (16 x 2048, f32) live in LDS (128KB; CDNA5 allows 320KB/WGP).
// V is pre-transposed per head ([b][h][d][t]) so P@V B-fragments are
// contiguous 16B loads along the K (= t) dimension.
// ---------------------------------------------------------------------------
__global__ __launch_bounds__(128) void attention_fused(
    const h16* __restrict__ q, const h16* __restrict__ k,
    const h16* __restrict__ vt, h16* __restrict__ ctx) {
  __shared__ float sc[16][SEQ];
  __shared__ float red[16][8];
  __shared__ float linv[16];

  const int tid  = threadIdx.x;
  const int wave = tid >> 5;  // 0..3
  const int lane = tid & 31;
  const int row  = lane & 15;
  const int hlf  = lane >> 4;
  const int m0   = hlf * 8;

  const int qt = blockIdx.x;  // 0..127 query tiles
  const int bh = blockIdx.y;  // 0..31  (b*16+h)

  const h16* qbase  = q + ((size_t)bh * SEQ + (size_t)qt * 16) * DK;
  const h16* kbase  = k + (size_t)bh * SEQ * DK;
  const h16* vtbase = vt + (size_t)bh * DK * SEQ;

  // Q fragments for this tile (K-dim 64 = two 32-chunks), reused all loop.
  v16h aq0 = load_frag(qbase, row, DK, 0, hlf);
  v16h aq1 = load_frag(qbase, row, DK, 32, hlf);

  // ---- scores: S = (Q K^T) / 8 ----
  for (int kt = wave; kt < SEQ / 16; kt += 4) {
    v16h bk0 = load_frag(kbase, kt * 16 + row, DK, 0, hlf);
    v16h bk1 = load_frag(kbase, kt * 16 + row, DK, 32, hlf);
    v8f s = {};
    s = wmma_f16(aq0, bk0, s);
    s = wmma_f16(aq1, bk1, s);
#pragma unroll
    for (int r = 0; r < 8; ++r) sc[m0 + r][kt * 16 + row] = s[r] * 0.125f;
  }
  __syncthreads();

  // ---- softmax over each row (8 threads per row) ----
  const int rr = tid >> 3;  // 0..15
  const int c0 = tid & 7;
  float mx = -3.0e38f;
  for (int c = c0; c < SEQ; c += 8) mx = fmaxf(mx, sc[rr][c]);
  red[rr][c0] = mx;
  __syncthreads();
  float m = -3.0e38f;
#pragma unroll
  for (int j = 0; j < 8; ++j) m = fmaxf(m, red[rr][j]);
  __syncthreads();
  float sum = 0.f;
  for (int c = c0; c < SEQ; c += 8) {
    float e = __expf(sc[rr][c] - m);
    sc[rr][c] = e;
    sum += e;
  }
  red[rr][c0] = sum;
  __syncthreads();
  float l = 0.f;
#pragma unroll
  for (int j = 0; j < 8; ++j) l += red[rr][j];
  linv[rr] = 1.0f / l;
  __syncthreads();

  // ---- O = P @ V ; each wave owns 16 of the 64 head dims ----
  v8f o = {};
  const h16* vrow = vtbase + (size_t)(wave * 16 + row) * SEQ;
  for (int kk = 0; kk < SEQ; kk += 32) {
    v16h a;
#pragma unroll
    for (int i = 0; i < 8; ++i) {
      a[i]     = (h16)sc[row][kk + m0 + i];
      a[i + 8] = (h16)sc[row][kk + 16 + m0 + i];
    }
    // B fragment: col n = head-dim (this wave's 16 dims), K = t, contiguous
    v16h bv = load_frag(vrow, 0, SEQ, kk, hlf);
    o = wmma_f16(a, bv, o);
  }

  const int b = bh >> 4;
  const int h = bh & 15;
  const int d = h * DK + wave * 16 + row;
#pragma unroll
  for (int r = 0; r < 8; ++r) {
    int grow = b * SEQ + qt * 16 + m0 + r;
    ctx[(size_t)grow * D_MODEL + d] = (h16)(o[r] * linv[m0 + r]);
  }
}

// ---------------------------------------------------------------------------
// Host-side pipeline (all launches on `stream`, graph-capture safe)
// ---------------------------------------------------------------------------
#define MB (1024ull * 1024ull)

extern "C" void kernel_launch(void* const* d_in, const int* in_sizes, int n_in,
                              void* d_out, int out_size, void* d_ws,
                              size_t ws_size, hipStream_t stream) {
  (void)in_sizes; (void)n_in; (void)out_size; (void)ws_size;

  const float* x   = (const float*)d_in[0];
  const float* wq  = (const float*)d_in[1];
  const float* bq  = (const float*)d_in[2];
  const float* wk  = (const float*)d_in[3];
  const float* bk  = (const float*)d_in[4];
  const float* wv  = (const float*)d_in[5];
  const float* bv  = (const float*)d_in[6];
  const float* wo  = (const float*)d_in[7];
  const float* bo  = (const float*)d_in[8];
  const float* w1  = (const float*)d_in[9];
  const float* b1  = (const float*)d_in[10];
  const float* w2  = (const float*)d_in[11];
  const float* b2  = (const float*)d_in[12];
  const float* g1  = (const float*)d_in[13];
  const float* be1 = (const float*)d_in[14];
  const float* g2  = (const float*)d_in[15];
  const float* be2 = (const float*)d_in[16];

  char* ws = (char*)d_ws;
  h16*   wqT  = (h16*)(ws + 0 * MB);    //  2 MB
  h16*   wkT  = (h16*)(ws + 2 * MB);    //  2 MB
  h16*   wvT  = (h16*)(ws + 4 * MB);    //  2 MB
  h16*   woT  = (h16*)(ws + 6 * MB);    //  2 MB
  h16*   w1T  = (h16*)(ws + 8 * MB);    //  8 MB  [4096][1024]
  h16*   w2T  = (h16*)(ws + 16 * MB);   //  8 MB  [1024][4096]
  h16*   n_h  = (h16*)(ws + 24 * MB);   //  8 MB  (reused for n1 and n2)
  h16*   q_h  = (h16*)(ws + 32 * MB);   //  8 MB  [b][h][t][d]
  h16*   k_h  = (h16*)(ws + 40 * MB);   //  8 MB  [b][h][t][d]
  h16*   vt_h = (h16*)(ws + 48 * MB);   //  8 MB  [b][h][d][t]  (transposed)
  h16*   ctx  = (h16*)(ws + 56 * MB);   //  8 MB  [b*t][1024]
  float* x1   = (float*)(ws + 64 * MB); // 16 MB
  h16*   h_h  = (h16*)(ws + 80 * MB);   // 32 MB  [4096][4096]

  // --- weight prep: f32 -> f16 transposed [N][K] ---
  const int nsq = D_MODEL * D_MODEL;
  const int nff = D_MODEL * DIM_FF;
  transpose_f32_to_f16<<<(nsq + 255) / 256, 256, 0, stream>>>(wq, wqT, D_MODEL, D_MODEL);
  transpose_f32_to_f16<<<(nsq + 255) / 256, 256, 0, stream>>>(wk, wkT, D_MODEL, D_MODEL);
  transpose_f32_to_f16<<<(nsq + 255) / 256, 256, 0, stream>>>(wv, wvT, D_MODEL, D_MODEL);
  transpose_f32_to_f16<<<(nsq + 255) / 256, 256, 0, stream>>>(wo, woT, D_MODEL, D_MODEL);
  transpose_f32_to_f16<<<(nff + 255) / 256, 256, 0, stream>>>(w1, w1T, D_MODEL, DIM_FF);
  transpose_f32_to_f16<<<(nff + 255) / 256, 256, 0, stream>>>(w2, w2T, DIM_FF, D_MODEL);

  // --- sublayer 1: LN -> QKV -> attention -> out-proj + residual ---
  layernorm_to_f16<<<M_ROWS, 256, 0, stream>>>(x, g1, be1, n_h);

  dim3 gsq(D_MODEL / 128, M_ROWS / 64);
  gemm_f16_wmma<MODE_QKV><<<gsq, 256, 0, stream>>>(n_h, wqT, bq, nullptr, q_h,
                                                   M_ROWS, D_MODEL, D_MODEL);
  gemm_f16_wmma<MODE_QKV><<<gsq, 256, 0, stream>>>(n_h, wkT, bk, nullptr, k_h,
                                                   M_ROWS, D_MODEL, D_MODEL);
  gemm_f16_wmma<MODE_V_T><<<gsq, 256, 0, stream>>>(n_h, wvT, bv, nullptr, vt_h,
                                                   M_ROWS, D_MODEL, D_MODEL);

  attention_fused<<<dim3(SEQ / 16, BATCH * N_HEADS), 128, 0, stream>>>(q_h, k_h,
                                                                       vt_h, ctx);

  gemm_f16_wmma<MODE_RES_F32><<<gsq, 256, 0, stream>>>(ctx, woT, bo, x, x1,
                                                       M_ROWS, D_MODEL, D_MODEL);

  // --- sublayer 2: LN -> FFN(GELU) + residual ---
  layernorm_to_f16<<<M_ROWS, 256, 0, stream>>>(x1, g2, be2, n_h);

  dim3 gff(DIM_FF / 128, M_ROWS / 64);
  gemm_f16_wmma<MODE_GELU_F16><<<gff, 256, 0, stream>>>(n_h, w1T, b1, nullptr,
                                                        h_h, M_ROWS, DIM_FF,
                                                        D_MODEL);

  gemm_f16_wmma<MODE_RES_F32><<<gsq, 256, 0, stream>>>(h_h, w2T, b2, x1,
                                                       (float*)d_out, M_ROWS,
                                                       D_MODEL, DIM_FF);
}
